// IntraFeatureAttention_85186381349585
// MI455X (gfx1250) — matrix operand order repaired and checked by reading, not compiled, not run on hardware
//
#include <hip/hip_runtime.h>
#include <stdint.h>

typedef __attribute__((ext_vector_type(16))) _Float16 v16h;
typedef __attribute__((ext_vector_type(8)))  _Float16 v8h;
typedef __attribute__((ext_vector_type(8)))  float    v8f;

#define DMODEL 768
#define NHEADS 12
#define DHEAD  64
#define MAXN   8192

// ---------------------------------------------------------------- fp32 -> f16
__global__ __launch_bounds__(256)
void cvt_f32_to_f16(const float* __restrict__ in, _Float16* __restrict__ out, int nElem) {
  int i = blockIdx.x * blockDim.x + threadIdx.x;
  int stride = gridDim.x * blockDim.x;
  for (; i < nElem; i += stride) out[i] = (_Float16)in[i];
}

// ------------------------------------------- prefix scan: query_mask -> idx[]
__global__ __launch_bounds__(1024)
void mask_scan(const int* __restrict__ qm, int n, int* __restrict__ idx) {
  __shared__ int cnt[1024];
  const int t = threadIdx.x;
  const int per = (n + 1023) >> 10;
  const int s = t * per;
  const int e = (s + per < n) ? (s + per) : n;
  int c = 0;
  for (int j = s; j < e; ++j) c += (qm[j] == 1) ? 1 : 0;
  cnt[t] = c;
  __syncthreads();
  for (int off = 1; off < 1024; off <<= 1) {
    int v = (t >= off) ? cnt[t - off] : 0;
    __syncthreads();
    cnt[t] += v;
    __syncthreads();
  }
  int base = cnt[t] - c;   // exclusive prefix
  for (int j = s; j < e; ++j)
    if (qm[j] == 1) idx[base++] = j;
}

// --------------------------------------------------------- WMMA f16 GEMM
// C[M,N] = A[M,K] * B[N,K]^T + bias[N]   (A rows optionally gathered by rowIdx)
// Block tile 128x128, BK=32, 256 threads = 8 waves (2 in M x 4 in N),
// wave tile 64x32 -> 4x2 v_wmma_f32_16x16x32_f16 accumulators.
// Tiles staged with global_load_async_to_lds_b128 (ASYNCcnt), double-buffered:
// one barrier per K-tile, next-tile copy overlaps current-tile WMMAs.
template<bool OUT_F16>
__global__ __launch_bounds__(256)
void gemm_wmma_f16(const _Float16* __restrict__ A, const int* __restrict__ rowIdx,
                   const _Float16* __restrict__ B, const float* __restrict__ bias,
                   void* __restrict__ Cout, int M, int N, int K)
{
  constexpr int BM = 128, BN = 128, BK = 32;
  __shared__ __align__(128) _Float16 sA[2][BM][BK];
  __shared__ __align__(128) _Float16 sB[2][BN][BK];

  const int tid  = threadIdx.x;
  const int lane = tid & 31;
  const int wv   = tid >> 5;
  const int wM   = wv >> 2;   // 0..1  (64 rows each)
  const int wN   = wv & 3;    // 0..3  (32 cols each)
  const int bn   = blockIdx.x * BN;
  const int bm   = blockIdx.y * BM;

  // staging map: thread t owns 32 bytes (16 f16) = half `lh` of tile row `lr`
  const int lr = tid >> 1;
  const int lh = tid & 1;
  const int aRowLog = bm + lr;
  const int aRow = rowIdx ? rowIdx[aRowLog] : aRowLog;
  const _Float16* aSrc = A + (size_t)aRow * K + lh * 16;
  const _Float16* bSrc = B + (size_t)(bn + lr) * K + lh * 16;

  // LDS byte offsets for the async-DMA destination (low 32 bits of the
  // generic pointer to shared memory are the LDS offset per the aperture map)
  const unsigned ldsA0 = (unsigned)(uintptr_t)&sA[0][lr][lh * 16];
  const unsigned ldsA1 = (unsigned)(uintptr_t)&sA[1][lr][lh * 16];
  const unsigned ldsB0 = (unsigned)(uintptr_t)&sB[0][lr][lh * 16];
  const unsigned ldsB1 = (unsigned)(uintptr_t)&sB[1][lr][lh * 16];

  auto stage = [&](int k0, int buf) {
    const unsigned la = buf ? ldsA1 : ldsA0;
    const unsigned lb = buf ? ldsB1 : ldsB0;
    const _Float16* ga = aSrc + k0;
    const _Float16* gb = bSrc + k0;
    asm volatile("global_load_async_to_lds_b128 %0, %1, off"
                 :: "v"(la),       "v"(ga)     : "memory");
    asm volatile("global_load_async_to_lds_b128 %0, %1, off"
                 :: "v"(la + 16u), "v"(ga + 8) : "memory");
    asm volatile("global_load_async_to_lds_b128 %0, %1, off"
                 :: "v"(lb),       "v"(gb)     : "memory");
    asm volatile("global_load_async_to_lds_b128 %0, %1, off"
                 :: "v"(lb + 16u), "v"(gb + 8) : "memory");
  };

  // fragment addressing per CDNA5 ISA layouts (wave32)
  const int lm   = lane & 15;
  const int khiA = (lane >> 4) * 8;    // A: lanes 0-15 hold K {0..7,16..23}, 16-31 hold {8..15,24..31}
  // B: lane = column, same K order as A: {0..7,16..23} / {8..15,24..31}

  v8f acc[4][2];
  #pragma unroll
  for (int i = 0; i < 4; ++i)
    #pragma unroll
    for (int j = 0; j < 2; ++j)
      acc[i][j] = (v8f){0.f,0.f,0.f,0.f,0.f,0.f,0.f,0.f};

  const int nk = K / BK;
  stage(0, 0);

  for (int kt = 0; kt < nk; ++kt) {
    const int buf = kt & 1;
    // own tile-kt copies done, then all waves' copies done / all prior reads retired
    asm volatile("s_wait_asynccnt 0" ::: "memory");
    __syncthreads();
    if (kt + 1 < nk) stage((kt + 1) * BK, buf ^ 1);
    if (kt + 2 < nk) {                 // keep L2 warm two tiles ahead
      __builtin_prefetch(aSrc + (kt + 2) * BK, 0, 0);
      __builtin_prefetch(bSrc + (kt + 2) * BK, 0, 0);
    }

    v16h af[4], bf[2];
    #pragma unroll
    for (int mt = 0; mt < 4; ++mt) {
      const int row = wM * 64 + mt * 16 + lm;
      v8h lo = *(const v8h*)&sA[buf][row][khiA];
      v8h hi = *(const v8h*)&sA[buf][row][16 + khiA];
      af[mt] = __builtin_shufflevector(lo, hi, 0,1,2,3,4,5,6,7,8,9,10,11,12,13,14,15);
    }
    #pragma unroll
    for (int nt = 0; nt < 2; ++nt) {
      const int col = wN * 32 + nt * 16 + lm;
      v8h lo = *(const v8h*)&sB[buf][col][khiA];
      v8h hi = *(const v8h*)&sB[buf][col][16 + khiA];
      bf[nt] = __builtin_shufflevector(lo, hi, 0,1,2,3,4,5,6,7,8,9,10,11,12,13,14,15);
    }
    #pragma unroll
    for (int mt = 0; mt < 4; ++mt)
      #pragma unroll
      for (int nt = 0; nt < 2; ++nt)
        acc[mt][nt] = __builtin_amdgcn_wmma_f32_16x16x32_f16(
            false, af[mt], false, bf[nt], (short)0, acc[mt][nt], false, false);
  }

  // epilogue: C/D layout -> lanes 0-15: M=vgpr, lanes 16-31: M=vgpr+8, N=lane&15
  #pragma unroll
  for (int mt = 0; mt < 4; ++mt) {
    const int rbase = bm + wM * 64 + mt * 16 + ((lane >> 4) << 3);
    #pragma unroll
    for (int nt = 0; nt < 2; ++nt) {
      const int col = bn + wN * 32 + nt * 16 + lm;
      const float bb = bias ? bias[col] : 0.f;
      v8f c = acc[mt][nt];
      #pragma unroll
      for (int r = 0; r < 8; ++r) {
        const float val = c[r] + bb;
        const size_t off = (size_t)(rbase + r) * N + col;
        if (OUT_F16) ((_Float16*)Cout)[off] = (_Float16)val;
        else         ((float*)Cout)[off]    = val;
      }
    }
  }
}

// ------------------------------------------------------------- attention
// grid (f, NHEADS), 128 threads. Segment softmax; fully-masked segments fall
// back to an unmasked softmax over all n keys (bias is uniform there).
__global__ __launch_bounds__(128)
void attention_kernel(const _Float16* __restrict__ Q, const _Float16* __restrict__ K,
                      const _Float16* __restrict__ V, const int* __restrict__ idx,
                      const int* __restrict__ ids, const int* __restrict__ padp,
                      _Float16* __restrict__ Aout, int f, int n)
{
  const int i = blockIdx.x;
  const int h = blockIdx.y;
  const int t = threadIdx.x;
  __shared__ float sc[MAXN];
  __shared__ float red[128];
  __shared__ float part[2][DHEAD];

  const int pad   = padp[0];
  const int left  = idx[i];
  const int right = (i + 1 < f) ? idx[i + 1] : n;
  const int seglen = right - left;

  // count valid (in-segment, non-pad) tokens
  int c = 0;
  for (int j = t; j < seglen; j += 128) c += (ids[left + j] != pad) ? 1 : 0;
  red[t] = (float)c;
  __syncthreads();
  for (int off = 64; off > 0; off >>= 1) { if (t < off) red[t] += red[t + off]; __syncthreads(); }
  const int nvalid = (int)red[0];
  __syncthreads();

  const _Float16* q = Q + (size_t)i * DMODEL + h * DHEAD;

  if (nvalid > 0) {
    // ---- segment path: masked scores underflow to exactly 0 after softmax
    float lmax = -1e30f;
    for (int j = t; j < seglen; j += 128) {
      const _Float16* kk = K + (size_t)(left + j) * DMODEL + h * DHEAD;
      float s = 0.f;
      #pragma unroll 8
      for (int d = 0; d < DHEAD; ++d) s += (float)q[d] * (float)kk[d];
      s *= 0.125f;
      if (ids[left + j] == pad) s = -1e30f;
      sc[j] = s;
      lmax = fmaxf(lmax, s);
    }
    red[t] = lmax; __syncthreads();
    for (int off = 64; off > 0; off >>= 1) { if (t < off) red[t] = fmaxf(red[t], red[t + off]); __syncthreads(); }
    const float m = red[0]; __syncthreads();

    float ls = 0.f;
    for (int j = t; j < seglen; j += 128) { float w = __expf(sc[j] - m); sc[j] = w; ls += w; }
    red[t] = ls; __syncthreads();
    for (int off = 64; off > 0; off >>= 1) { if (t < off) red[t] += red[t + off]; __syncthreads(); }
    const float inv = 1.f / red[0];
    __syncthreads();

    for (int d = t; d < DHEAD; d += 128) {
      float o = 0.f;
      for (int j = 0; j < seglen; ++j)
        o += sc[j] * (float)V[(size_t)(left + j) * DMODEL + h * DHEAD + d];
      Aout[(size_t)i * DMODEL + h * DHEAD + d] = (_Float16)(o * inv);
    }
  } else {
    // ---- fully-masked segment: uniform bias -> plain softmax over ALL n keys
    float lmax = -1e30f;
    for (int j = t; j < n; j += 128) {
      const _Float16* kk = K + (size_t)j * DMODEL + h * DHEAD;
      float s = 0.f;
      #pragma unroll 8
      for (int d = 0; d < DHEAD; ++d) s += (float)q[d] * (float)kk[d];
      s *= 0.125f;
      sc[j] = s;
      lmax = fmaxf(lmax, sc[j]);
    }
    red[t] = lmax; __syncthreads();
    for (int off = 64; off > 0; off >>= 1) { if (t < off) red[t] = fmaxf(red[t], red[t + off]); __syncthreads(); }
    const float m = red[0]; __syncthreads();

    float ls = 0.f;
    for (int j = t; j < n; j += 128) { float w = __expf(sc[j] - m); sc[j] = w; ls += w; }
    red[t] = ls; __syncthreads();
    for (int off = 64; off > 0; off >>= 1) { if (t < off) red[t] += red[t + off]; __syncthreads(); }
    const float S = red[0]; __syncthreads();

    const int d = t & 63;
    const int g = t >> 6;   // 2 key groups
    float o = 0.f;
    for (int j = g; j < n; j += 2)
      o += sc[j] * (float)V[(size_t)j * DMODEL + h * DHEAD + d];
    part[g][d] = o;
    __syncthreads();
    if (t < DHEAD)
      Aout[(size_t)i * DMODEL + h * DHEAD + t] = (_Float16)((part[0][t] + part[1][t]) / S);
  }
}

// ---------------------------------------------------------------- launch
extern "C" void kernel_launch(void* const* d_in, const int* in_sizes, int n_in,
                              void* d_out, int out_size, void* d_ws, size_t ws_size,
                              hipStream_t stream)
{
  (void)n_in; (void)ws_size;
  const float* x_qk = (const float*)d_in[0];
  const float* x_v  = (const float*)d_in[1];
  const int*   qm   = (const int*)d_in[2];
  const int*   ids  = (const int*)d_in[3];
  const int*   padp = (const int*)d_in[4];
  const float* Wq = (const float*)d_in[5];
  const float* bq = (const float*)d_in[6];
  const float* Wk = (const float*)d_in[7];
  const float* bk = (const float*)d_in[8];
  const float* Wv = (const float*)d_in[9];
  const float* bv = (const float*)d_in[10];
  const float* Wo = (const float*)d_in[11];
  const float* bo = (const float*)d_in[12];
  float* out = (float*)d_out;

  const int D = DMODEL;
  const int n = in_sizes[0] / D;     // 8192 tokens
  const int f = out_size   / D;      // 1024 features

  // workspace carve (256B aligned); total ~58 MB
  char* p = (char*)d_ws;
  auto carve = [&](size_t bytes) -> void* {
    void* r = (void*)p;
    p += (bytes + 255) & ~(size_t)255;
    return r;
  };
  _Float16* Xqk16 = (_Float16*)carve((size_t)n * D * 2);
  _Float16* Xv16  = (_Float16*)carve((size_t)n * D * 2);
  _Float16* Wq16  = (_Float16*)carve((size_t)D * D * 2);
  _Float16* Wk16  = (_Float16*)carve((size_t)D * D * 2);
  _Float16* Wv16  = (_Float16*)carve((size_t)D * D * 2);
  _Float16* Wo16  = (_Float16*)carve((size_t)D * D * 2);
  _Float16* K16   = (_Float16*)carve((size_t)n * D * 2);
  _Float16* V16   = (_Float16*)carve((size_t)n * D * 2);
  _Float16* Q16   = (_Float16*)carve((size_t)f * D * 2);
  _Float16* A16   = (_Float16*)carve((size_t)f * D * 2);
  int*      idx   = (int*)carve((size_t)f * 4);

  // 1) precision downconvert
  cvt_f32_to_f16<<<512, 256, 0, stream>>>(x_qk, Xqk16, n * D);
  cvt_f32_to_f16<<<512, 256, 0, stream>>>(x_v,  Xv16,  n * D);
  cvt_f32_to_f16<<<128, 256, 0, stream>>>(Wq, Wq16, D * D);
  cvt_f32_to_f16<<<128, 256, 0, stream>>>(Wk, Wk16, D * D);
  cvt_f32_to_f16<<<128, 256, 0, stream>>>(Wv, Wv16, D * D);
  cvt_f32_to_f16<<<128, 256, 0, stream>>>(Wo, Wo16, D * D);

  // 2) feature index scan
  mask_scan<<<1, 1024, 0, stream>>>(qm, n, idx);

  // 3) projections (WMMA, async-LDS pipelined)
  dim3 gKV(D / 128, n / 128);   // 6 x 64
  dim3 gF (D / 128, f / 128);   // 6 x 8
  gemm_wmma_f16<true ><<<gKV, 256, 0, stream>>>(Xqk16, nullptr, Wk16, bk, K16, n, D, D);
  gemm_wmma_f16<true ><<<gKV, 256, 0, stream>>>(Xv16,  nullptr, Wv16, bv, V16, n, D, D);
  gemm_wmma_f16<true ><<<gF,  256, 0, stream>>>(Xqk16, idx,     Wq16, bq, Q16, f, D, D);

  // 4) block-diagonal attention (+ global fallback for fully-padded segments)
  attention_kernel<<<dim3(f, NHEADS), 128, 0, stream>>>(Q16, K16, V16, idx, ids, padp, A16, f, n);

  // 5) output projection straight to f32 d_out
  gemm_wmma_f16<false><<<gF, 256, 0, stream>>>(A16, nullptr, Wo16, bo, out, f, D, D);
}
